// QuaternionAttention_7361573945756
// MI455X (gfx1250) — compile-verified, matrix operand order checked by reading, not run on hardware
//
#include <hip/hip_runtime.h>
#include <hip/hip_bf16.h>

// ---------------------------------------------------------------------------
// QuaternionAttention on MI455X (gfx1250): bf16 WMMA (16x16x32, f32 accum)
// with double-buffered async global->LDS tile staging (ASYNCcnt path),
// LDS-cached row softmax, and the degenerate final einsum (rowsum * V).
// ---------------------------------------------------------------------------

typedef __attribute__((ext_vector_type(16))) __bf16 v16bf;
typedef __attribute__((ext_vector_type(8)))  __bf16 v8bf;
typedef __attribute__((ext_vector_type(8)))  float  v8f;

// ---------------- f32 -> bf16 conversion (grid-stride) ---------------------
__global__ __launch_bounds__(256) void qa_cvt_bf16(const float* __restrict__ in,
                                                   __bf16* __restrict__ out,
                                                   long n) {
  long i = (long)blockIdx.x * blockDim.x + threadIdx.x;
  long stride = (long)gridDim.x * blockDim.x;
  for (; i < n; i += stride) out[i] = (__bf16)in[i];
}

// ---------------- CDNA5 async global->LDS 16B/lane copy --------------------
__device__ __forceinline__ void qa_async_ld16(unsigned lds_addr,
                                              unsigned long long gaddr) {
  asm volatile("global_load_async_to_lds_b128 %0, %1, off"
               :: "v"(lds_addr), "v"(gaddr) : "memory");
}

// ---------------- WMMA fragment from LDS (A 16x32 / B 32x16 bf16) ----------
// Lane layout (ISA 7.12.2): lane = row(A)/col(B) in 0..15, hi = lane>>4.
// Elements 0..7  <- K = hi*8 + 0..7      (bytes p+0  .. p+15)
// Elements 8..15 <- K = 16 + hi*8 + 0..7 (bytes p+32 .. p+47)
__device__ __forceinline__ v16bf qa_lds_frag(const char* p) {
  v8bf lo = *(const v8bf*)(p);
  v8bf hi = *(const v8bf*)(p + 32);
  v16bf r;
#pragma unroll
  for (int i = 0; i < 8; ++i) { r[i] = lo[i]; r[i + 8] = hi[i]; }
  return r;
}

// ---------------- NT GEMM: C[m,n] = scale*(sum_k A[m,k]*Bm[n,k] + bias[n]) -
// Block: 256 threads = 8 waves; tile 128(M) x 64(N); wave tile 32x32
// (2x2 fragments of v_wmma_f32_16x16x32_bf16). lda = ldb = K, ldc = N.
// A/B tiles staged to LDS with async b128 loads, double buffered.
__global__ __launch_bounds__(256)
void qa_gemm_nt_bf16(const __bf16* __restrict__ A, const __bf16* __restrict__ Bm,
                     const float* __restrict__ bias,
                     float* __restrict__ outF, __bf16* __restrict__ outB,
                     int N, int K, float scale,
                     long strideA, long strideB, long strideC) {
  constexpr int AP     = 80;            // padded LDS row stride (64B data + 16B)
  constexpr int ABYTES = 128 * AP;      // 10240 B per A buffer
  constexpr int BBYTES = 64 * AP;       // 5120 B per B buffer
  __shared__ __align__(16) char smem[2 * (ABYTES + BBYTES)];

  const int bz = blockIdx.z;
  A  += (long)bz * strideA;
  Bm += (long)bz * strideB;

  const int tile_n0 = blockIdx.x * 64;
  const int tile_m0 = blockIdx.y * 128;
  const int t    = threadIdx.x;
  const int wave = t >> 5;
  const int lane = t & 31;
  const int l15  = lane & 15;
  const int lhi  = lane >> 4;
  const int wm   = (wave >> 1) * 32;    // wave M offset inside tile
  const int wn   = (wave & 1) * 32;     // wave N offset inside tile

  const unsigned smem0  = (unsigned)(unsigned long long)(void*)&smem[0];
  const unsigned aOff[2] = { 0u, (unsigned)(ABYTES + BBYTES) };
  const unsigned bOff[2] = { (unsigned)ABYTES, (unsigned)(2 * ABYTES + BBYTES) };

  // Staging assignment: A tile 128 rows x 64B -> 32B per thread (2 async b128)
  //                     B tile  64 rows x 64B -> 16B per thread (1 async b128)
  const int arow = t >> 1, acol = (t & 1) * 32;
  const int brow = t >> 2, bcol = (t & 3) * 16;
  const unsigned long long agBase =
      (unsigned long long)(const char*)(A + (size_t)(tile_m0 + arow) * K) + acol;
  const unsigned long long bgBase =
      (unsigned long long)(const char*)(Bm + (size_t)(tile_n0 + brow) * K) + bcol;
  const unsigned aLds = (unsigned)(arow * AP + acol);
  const unsigned bLds = (unsigned)(brow * AP + bcol);

  auto issue = [&](int buf, int k0) {
    const unsigned long long kb = (unsigned long long)k0 * 2;  // bytes
    qa_async_ld16(smem0 + aOff[buf] + aLds,      agBase + kb);
    qa_async_ld16(smem0 + aOff[buf] + aLds + 16, agBase + kb + 16);
    qa_async_ld16(smem0 + bOff[buf] + bLds,      bgBase + kb);
  };

  const float bv0 = bias ? bias[tile_n0 + wn + l15]      : 0.0f;
  const float bv1 = bias ? bias[tile_n0 + wn + 16 + l15] : 0.0f;
  v8f acc00, acc01, acc10, acc11;
#pragma unroll
  for (int r = 0; r < 8; ++r) {
    acc00[r] = bv0; acc01[r] = bv1; acc10[r] = bv0; acc11[r] = bv1;
  }

  issue(0, 0);
  for (int k0 = 0; k0 < K; k0 += 32) {
    const int cur = (k0 >> 5) & 1;
    if (k0 + 32 < K) {
      issue(cur ^ 1, k0 + 32);                       // prefetch next buffer
      asm volatile("s_wait_asynccnt 0x3" ::: "memory");  // current buffer done
    } else {
      asm volatile("s_wait_asynccnt 0x0" ::: "memory");
    }
    __syncthreads();

    const char* sa = smem + aOff[cur];
    const char* sb = smem + bOff[cur];
    v16bf a0 = qa_lds_frag(sa + (wm + l15) * AP + lhi * 16);
    v16bf a1 = qa_lds_frag(sa + (wm + 16 + l15) * AP + lhi * 16);
    v16bf b0 = qa_lds_frag(sb + (wn + l15) * AP + lhi * 16);
    v16bf b1 = qa_lds_frag(sb + (wn + 16 + l15) * AP + lhi * 16);

    acc00 = __builtin_amdgcn_wmma_f32_16x16x32_bf16(false, a0, false, b0,
                                                    (short)0, acc00, false, false);
    acc01 = __builtin_amdgcn_wmma_f32_16x16x32_bf16(false, a0, false, b1,
                                                    (short)0, acc01, false, false);
    acc10 = __builtin_amdgcn_wmma_f32_16x16x32_bf16(false, a1, false, b0,
                                                    (short)0, acc10, false, false);
    acc11 = __builtin_amdgcn_wmma_f32_16x16x32_bf16(false, a1, false, b1,
                                                    (short)0, acc11, false, false);
    __syncthreads();
  }

  // C/D layout: lane -> col n = lane&15; VGPR r -> row m = 8*(lane>>4) + r.
  const int m_base = tile_m0 + wm;
  const int n_base = tile_n0 + wn;
  const size_t cbase = (size_t)bz * (size_t)strideC;
#pragma unroll
  for (int r = 0; r < 8; ++r) {
    const int m0 = m_base + 8 * lhi + r;
    const int m1 = m0 + 16;
    const int n0 = n_base + l15;
    const int n1 = n0 + 16;
    const float v00 = acc00[r] * scale, v01 = acc01[r] * scale;
    const float v10 = acc10[r] * scale, v11 = acc11[r] * scale;
    if (outF) {
      float* c = outF + cbase;
      c[(size_t)m0 * N + n0] = v00; c[(size_t)m0 * N + n1] = v01;
      c[(size_t)m1 * N + n0] = v10; c[(size_t)m1 * N + n1] = v11;
    }
    if (outB) {
      __bf16* c = outB + cbase;
      c[(size_t)m0 * N + n0] = (__bf16)v00; c[(size_t)m0 * N + n1] = (__bf16)v01;
      c[(size_t)m1 * N + n0] = (__bf16)v10; c[(size_t)m1 * N + n1] = (__bf16)v11;
    }
  }
}

// ---------------- Row softmax, row cached in LDS (4096 f32 = 16KB) ---------
__global__ __launch_bounds__(256)
void qa_softmax_row(float* __restrict__ w, float* __restrict__ rowsum, int S) {
  __shared__ float row[4096];
  __shared__ float red[256];
  const int tid = threadIdx.x;
  const size_t base = (size_t)blockIdx.x * S;

  float m = -__builtin_huge_valf();
  for (int i = tid; i < S; i += 256) {
    float v = w[base + i];
    row[i] = v;
    m = fmaxf(m, v);
  }
  red[tid] = m; __syncthreads();
#pragma unroll
  for (int s = 128; s > 0; s >>= 1) {
    if (tid < s) red[tid] = fmaxf(red[tid], red[tid + s]);
    __syncthreads();
  }
  const float rowmax = red[0];
  __syncthreads();

  float sum = 0.0f;
  for (int i = tid; i < S; i += 256) {
    float e = __expf(row[i] - rowmax);
    row[i] = e;
    sum += e;
  }
  red[tid] = sum; __syncthreads();
#pragma unroll
  for (int s = 128; s > 0; s >>= 1) {
    if (tid < s) red[tid] += red[tid + s];
    __syncthreads();
  }
  const float inv = 1.0f / red[0];
  __syncthreads();

  float rs = 0.0f;
  for (int i = tid; i < S; i += 256) {
    float e = row[i] * inv;
    w[base + i] = e;
    rs += e;
  }
  red[tid] = rs; __syncthreads();
#pragma unroll
  for (int s = 128; s > 0; s >>= 1) {
    if (tid < s) red[tid] += red[tid + s];
    __syncthreads();
  }
  if (tid == 0) rowsum[blockIdx.x] = red[0];
}

// ---------------- out[b,s,d] = rowsum[b,s] * V[b,s,d]  (d < 512) -----------
__global__ __launch_bounds__(256)
void qa_scale_v(const float* __restrict__ V, const float* __restrict__ rowsum,
                float* __restrict__ out, long n) {
  long i = (long)blockIdx.x * blockDim.x + threadIdx.x;
  long stride = (long)gridDim.x * blockDim.x;
  for (; i < n; i += stride) out[i] = V[i] * rowsum[i >> 9];  // D = 512
}

// ---------------------------------------------------------------------------
extern "C" void kernel_launch(void* const* d_in, const int* in_sizes, int n_in,
                              void* d_out, int out_size, void* d_ws, size_t ws_size,
                              hipStream_t stream) {
  (void)in_sizes; (void)n_in; (void)out_size; (void)ws_size;
  constexpr int  Bn = 4, S = 4096, E = 128, D = 512;
  constexpr long M  = (long)Bn * S;       // 16384 rows
  constexpr long NA = M * D;              // 8,388,608 activation elements
  constexpr long NW = (long)D * D;        // 262,144 weight elements

  const float* q  = (const float*)d_in[0];
  const float* k  = (const float*)d_in[1];
  const float* v  = (const float*)d_in[2];
  const float* Wq = (const float*)d_in[3];
  const float* bq = (const float*)d_in[4];
  const float* Wk = (const float*)d_in[5];
  const float* bk = (const float*)d_in[6];
  const float* Wv = (const float*)d_in[7];
  const float* bv = (const float*)d_in[8];

  char* ws = (char*)d_ws;
  size_t off = 0;
  auto take = [&](size_t bytes) -> char* {
    char* p = ws + off;
    off = (off + bytes + 255) & ~(size_t)255;
    return p;
  };
  __bf16* qb  = (__bf16*)take(NA * 2);
  __bf16* kb  = (__bf16*)take(NA * 2);
  __bf16* vb  = (__bf16*)take(NA * 2);
  __bf16* Wqb = (__bf16*)take(NW * 2);
  __bf16* Wkb = (__bf16*)take(NW * 2);
  __bf16* Wvb = (__bf16*)take(NW * 2);
  __bf16* Qb  = (__bf16*)take(NA * 2);
  __bf16* Kb  = (__bf16*)take(NA * 2);
  float*  Vf  = (float*)take(NA * 4);
  float*  rsm = (float*)take(M * 4);

  float* out_v = (float*)d_out;          // [B,S,E,4] = NA floats
  float* out_w = out_v + NA;             // [B,S,S]   weights

  // 1. conversions
  qa_cvt_bf16<<<4096, 256, 0, stream>>>(q, qb, NA);
  qa_cvt_bf16<<<4096, 256, 0, stream>>>(k, kb, NA);
  qa_cvt_bf16<<<4096, 256, 0, stream>>>(v, vb, NA);
  qa_cvt_bf16<<<512,  256, 0, stream>>>(Wq, Wqb, NW);
  qa_cvt_bf16<<<512,  256, 0, stream>>>(Wk, Wkb, NW);
  qa_cvt_bf16<<<512,  256, 0, stream>>>(Wv, Wvb, NW);

  // 2. projections: [16384x512] @ [512x512]^T (+bias)
  dim3 gp(D / 64, M / 128, 1);
  qa_gemm_nt_bf16<<<gp, 256, 0, stream>>>(qb, Wqb, bq, nullptr, Qb,
                                          D, D, 1.0f, 0, 0, 0);
  qa_gemm_nt_bf16<<<gp, 256, 0, stream>>>(kb, Wkb, bk, nullptr, Kb,
                                          D, D, 1.0f, 0, 0, 0);
  qa_gemm_nt_bf16<<<gp, 256, 0, stream>>>(vb, Wvb, bv, Vf, nullptr,
                                          D, D, 1.0f, 0, 0, 0);

  // 3. scores = Q @ K^T / sqrt(E), batched over B (grid.z)
  const float inv_sqrt_e = 0.08838834764831845f;  // 1/sqrt(128)
  dim3 gs(S / 64, S / 128, Bn);
  qa_gemm_nt_bf16<<<gs, 256, 0, stream>>>(Qb, Kb, nullptr, out_w, nullptr,
                                          S, D, inv_sqrt_e,
                                          (long)S * D, (long)S * D, (long)S * S);

  // 4. softmax rows in place (+ faithful rowsums)
  qa_softmax_row<<<(int)M, 256, 0, stream>>>(out_w, rsm, S);

  // 5. degenerate einsum: out = rowsum(weights) * V
  qa_scale_v<<<8192, 256, 0, stream>>>(Vf, rsm, out_v, NA);
}
